// TransformerLiftSplatShoot_8504035246248
// MI455X (gfx1250) — matrix operand order repaired and checked by reading, not compiled, not run on hardware
//
#include <hip/hip_runtime.h>
#include <hip/hip_bf16.h>

// ---------------- problem constants (from reference) ----------------
#define BATCH   8
#define FH      64
#define FW      64
#define HW      (FH * FW)          // 4096
#define NPIX    (BATCH * HW)       // 32768
#define IN_CH   256
#define NDEPTH  49
#define NCTX    64
#define OCH     (NDEPTH + NCTX)    // 113
#define OCH_PAD 128
#define NX0     100                // x bins
#define NX2     98                 // z bins
#define PWS_ELEMS (BATCH * NX2 * NX0 * NCTX)   // 5,017,600

typedef __attribute__((ext_vector_type(16))) _Float16 v16h;
typedef __attribute__((ext_vector_type(8)))  float    v8f;

// ---------------------------------------------------------------
// Kernel 0: zero the pooled scatter buffer (graph replays; harness
// does not re-poison, and we accumulate with atomics).
// ---------------------------------------------------------------
__global__ void lss_zero(float* __restrict__ buf, int n) {
    int i = blockIdx.x * blockDim.x + threadIdx.x;
    int stride = gridDim.x * blockDim.x;
    for (; i < n; i += stride) buf[i] = 0.0f;
}

// ---------------------------------------------------------------
// Kernel 1: depthnet 1x1 conv as WMMA GEMM.
//   feat[o, p] = sum_k W[o,k] * F[b, k, hw] + bias[o]
//   O = 113 (8 mtiles of 16), P = 32768, K = 256.
// One wave computes a 16x16 (o x p) tile; 8 waves per block cover
// all 8 o-tiles of the same 16 pixels (feature reuse in L0/WGP$).
// Output stored pixel-major padded: ws_feat[p*128 + o].
// ---------------------------------------------------------------
__global__ __launch_bounds__(256) void lss_depthnet_wmma(
    const float* __restrict__ feats,   // [B, 256, H, W]
    const float* __restrict__ dw,      // [113, 256]
    const float* __restrict__ dbias,   // [113]
    float* __restrict__ ws_feat)       // [NPIX, 128]
{
    const int lane  = threadIdx.x & 31;
    const int mtile = threadIdx.x >> 5;     // 0..7 -> output channels mtile*16..+15
    const int ntile = blockIdx.x;           // 0..2047 -> pixels ntile*16..+15

    // ---- per-lane A-fragment coordinates (16x32 f16 layout, ISA 7.12.2) ----
    const int am = lane & 15;               // matrix row within tile
    const int o_row = mtile * 16 + am;      // output channel this lane feeds A with
    const int kb = (lane >> 4) << 3;        // 0 or 8: K sub-block base

    // ---- per-lane B-fragment coordinates (32x16 f16 layout) ----
    const int bn   = lane & 15;
    const int p    = ntile * 16 + bn;       // global pixel
    const int b    = p >> 12;               // / 4096
    const int hw   = p & 4095;
    const int koff = (lane >> 4) << 4;      // 0 or 16
    const float* fcol = feats + (size_t)b * IN_CH * HW + hw;

    v8f acc = {};

    #pragma unroll
    for (int k0 = 0; k0 < IN_CH; k0 += 32) {
        // A: weights tile, rows = output channels, cols = k
        v16h a;
        if (o_row < OCH) {
            const float* wrow = dw + (size_t)o_row * IN_CH + k0;
            #pragma unroll
            for (int e = 0; e < 8; ++e) {
                a[e]     = (_Float16)wrow[kb + e];        // K = kb + e
                a[8 + e] = (_Float16)wrow[kb + 16 + e];   // K = kb + 16 + e
            }
        } else {
            #pragma unroll
            for (int e = 0; e < 16; ++e) a[e] = (_Float16)0.0f;
        }

        // B: feature tile, rows = k, cols = pixels
        v16h bm;
        #pragma unroll
        for (int e = 0; e < 16; ++e)
            bm[e] = (_Float16)fcol[(size_t)(k0 + (koff >> 1) + (e & 7) + ((e >> 3) << 4)) * HW];

        acc = __builtin_amdgcn_wmma_f32_16x16x32_f16(
            /*neg_a=*/false, a, /*neg_b=*/false, bm,
            /*c_mod=*/(short)0, acc, /*reuse_a=*/false, /*reuse_b=*/false);
    }

    // ---- C layout: VGPR r, lanes 0-15 -> M=r, lanes 16-31 -> M=8+r, N=lane%16 ----
    const int obase = mtile * 16 + ((lane >> 4) << 3);   // 8 consecutive channels
    float v[8];
    #pragma unroll
    for (int r = 0; r < 8; ++r) {
        int o = obase + r;
        v[r] = acc[r] + ((o < OCH) ? dbias[o] : 0.0f);
    }
    float4* dst = (float4*)(ws_feat + (size_t)p * OCH_PAD + obase);
    dst[0] = make_float4(v[0], v[1], v[2], v[3]);
    dst[1] = make_float4(v[4], v[5], v[6], v[7]);
}

// ---------------------------------------------------------------
// Kernel 2: softmax over 49 depth logits + lift + voxel scatter.
// 64 lanes per pixel (lane == BEV channel), 4 pixels per block.
// Scatter target is channel-CONTIGUOUS pooled buffer pws[b][z][x][c]
// so each atomic wavefront touches 256 contiguous bytes (2 lines)
// instead of 64 scattered lines -> ~8x fewer L2 atomic transactions.
// ---------------------------------------------------------------
__global__ __launch_bounds__(256) void lss_lift_splat(
    const float* __restrict__ ws_feat,  // [NPIX, 128]
    const float* __restrict__ kmat,     // [B, 3, 3]
    float* __restrict__ pws)            // [B, 98, 100, 64]
{
    __shared__ float s_log[4][64];
    __shared__ float s_exp[4][64];

    const int tid = threadIdx.x;
    const int g      = tid >> 6;        // pixel group in block: 0..3
    const int lane64 = tid & 63;        // BEV channel
    const int p  = blockIdx.x * 4 + g;
    const int b  = p >> 12;
    const int hw = p & 4095;
    const int h  = hw >> 6;
    const int w  = hw & 63;

    const float* base = ws_feat + (size_t)p * OCH_PAD;
    if (lane64 < NDEPTH) s_log[g][lane64] = base[lane64];
    const float ctx = base[NDEPTH + lane64];
    __syncthreads();

    float mx = -3.402823466e38f;
    #pragma unroll 7
    for (int d = 0; d < NDEPTH; ++d) mx = fmaxf(mx, s_log[g][d]);
    if (lane64 < NDEPTH) s_exp[g][lane64] = __expf(s_log[g][lane64] - mx);
    __syncthreads();

    float denom = 0.0f;
    #pragma unroll 7
    for (int d = 0; d < NDEPTH; ++d) denom += s_exp[g][d];
    const float rdenom = 1.0f / denom;

    // per-batch K^-1 (closed form 3x3)
    const float* K = kmat + b * 9;
    const float a0 = K[0], a1 = K[1], a2 = K[2];
    const float a3 = K[3], a4 = K[4], a5 = K[5];
    const float a6 = K[6], a7 = K[7], a8 = K[8];
    const float det = a0*(a4*a8 - a5*a7) - a1*(a3*a8 - a5*a6) + a2*(a3*a7 - a4*a6);
    const float id  = 1.0f / det;
    const float i00 = (a4*a8 - a5*a7)*id, i01 = (a2*a7 - a1*a8)*id, i02 = (a1*a5 - a2*a4)*id;
    const float i10 = (a5*a6 - a3*a8)*id, i11 = (a0*a8 - a2*a6)*id, i12 = (a2*a3 - a0*a5)*id;
    const float i20 = (a3*a7 - a4*a6)*id, i21 = (a1*a6 - a0*a7)*id, i22 = (a0*a4 - a1*a3)*id;

    // pixel coords: linspace(0, 255, 64)
    const float u = (float)w * (255.0f / 63.0f);
    const float v = (float)h * (255.0f / 63.0f);

    const size_t bbase = (size_t)b * (NX2 * NX0 * NCTX);

    for (int d = 0; d < NDEPTH; ++d) {
        const float dep = 1.0f + (float)d;          // depth bins 1..49
        const float px = u * dep, py = v * dep, pz = dep;
        const float gx = i00*px + i01*py + i02*pz;
        const float gy = i10*px + i11*py + i12*pz;
        const float gz = i20*px + i21*py + i22*pz;
        // (geom - bxmin)/dx, trunc toward zero (matches astype(int32))
        const int cx = (int)((gx + 25.0f) / 0.5f);
        const int cy = (int)((gy + 10.0f) / 20.0f);
        const int cz = (int)((gz - 1.0f)  / 0.5f);
        if (cx >= 0 && cx < NX0 && cy >= 0 && cy < 1 && cz >= 0 && cz < NX2) {
            const float val = s_exp[g][d] * rdenom * ctx;
            atomicAdd(pws + bbase + ((size_t)cz * NX0 + cx) * NCTX + lane64, val);
        }
    }
}

// ---------------------------------------------------------------
// Kernel 3: transpose (B, z, x, C) -> (B, C, z, x) == final output.
// One block per (b, z) slice: 100 x * 64 c = 6400 floats via a
// padded LDS tile (stride 65 -> bank-conflict-free both phases).
// Both global read and write are coalesced; writes every output
// element, so no separate d_out zeroing is needed.
// ---------------------------------------------------------------
__global__ __launch_bounds__(256) void lss_transpose(
    const float* __restrict__ pws,      // [B, 98, 100, 64]
    float* __restrict__ out)            // [B, 64, 98, 100]
{
    __shared__ float tile[NX0 * 65];    // [x][c] padded

    const int bz = blockIdx.x;          // 0..783
    const int b  = bz / NX2;
    const int z  = bz - b * NX2;

    const float* src = pws + (size_t)bz * (NX0 * NCTX);
    for (int e = threadIdx.x; e < NX0 * NCTX; e += 256) {
        int x = e >> 6, c = e & 63;
        tile[x * 65 + c] = src[e];
    }
    __syncthreads();
    for (int e = threadIdx.x; e < NX0 * NCTX; e += 256) {
        int c = e / NX0, x = e - c * NX0;
        out[(((size_t)b * NCTX + c) * NX2 + z) * NX0 + x] = tile[x * 65 + c];
    }
}

// ---------------------------------------------------------------
// Launch
// ---------------------------------------------------------------
extern "C" void kernel_launch(void* const* d_in, const int* in_sizes, int n_in,
                              void* d_out, int out_size, void* d_ws, size_t ws_size,
                              hipStream_t stream) {
    (void)in_sizes; (void)n_in; (void)ws_size; (void)out_size;
    const float* feats = (const float*)d_in[0];   // [8,256,64,64]
    const float* kmat  = (const float*)d_in[1];   // [8,3,3]
    const float* dw    = (const float*)d_in[2];   // [113,256]
    const float* dbias = (const float*)d_in[3];   // [113]
    float* out = (float*)d_out;                   // [8,64,98,100] = 5,017,600

    float* ws_feat = (float*)d_ws;                               // 16.78 MB
    float* pws     = (float*)((char*)d_ws + (size_t)NPIX * OCH_PAD * sizeof(float)); // 20.07 MB

    lss_zero<<<4096, 256, 0, stream>>>(pws, PWS_ELEMS);
    lss_depthnet_wmma<<<NPIX / 16, 256, 0, stream>>>(feats, dw, dbias, ws_feat);
    lss_lift_splat<<<NPIX / 4, 256, 0, stream>>>(ws_feat, kmat, pws);
    lss_transpose<<<BATCH * NX2, 256, 0, stream>>>(pws, out);
}
